// LearnedRewiring_68410239090697
// MI455X (gfx1250) — matrix operand order repaired and checked by reading, not compiled, not run on hardware
//
#include <hip/hip_runtime.h>
#include <hip/hip_bf16.h>

typedef __attribute__((ext_vector_type(16))) _Float16 v16h;
typedef __attribute__((ext_vector_type(8)))  _Float16 v8h;
typedef __attribute__((ext_vector_type(8)))  float    v8f;
typedef __attribute__((ext_vector_type(4)))  float    f4;

#define TDIM 2048
#define CDIM 1024
#define HNUM 8
#define RDIM 32
#define PROJC (HNUM * RDIM)   // 256
#define BROWS (2 * TDIM)      // 4096 rows of x / sq / sk

// ---------------------------------------------------------------------------
// Prep: cast x (f32 -> f16), row-major [4096,1024]; 8 elements per thread.
// x is read exactly once -> non-temporal load. xh is re-read 32x by lr_proj
// -> regular (cache-resident) store.
__global__ void lr_cast_x(const float* __restrict__ x, _Float16* __restrict__ xh, int n8) {
    int i = blockIdx.x * blockDim.x + threadIdx.x;
    if (i >= n8) return;
    const f4* src = (const f4*)(x + (size_t)i * 8);
    f4 a = __builtin_nontemporal_load(src);
    f4 b = __builtin_nontemporal_load(src + 1);
    v8h o;
    o[0] = (_Float16)a.x; o[1] = (_Float16)a.y; o[2] = (_Float16)a.z; o[3] = (_Float16)a.w;
    o[4] = (_Float16)b.x; o[5] = (_Float16)b.y; o[6] = (_Float16)b.z; o[7] = (_Float16)b.w;
    *(v8h*)(xh + (size_t)i * 8) = o;
}

// Prep: transpose+cast Wq,Wk [1024,256] f32 -> [256,1024] f16
__global__ void lr_transpose_w(const float* __restrict__ Wq, const float* __restrict__ Wk,
                               _Float16* __restrict__ wqt, _Float16* __restrict__ wkt) {
    int idx = blockIdx.x * blockDim.x + threadIdx.x;  // over 256*1024
    if (idx >= PROJC * CDIM) return;
    int c = idx >> 10;          // output row  (0..255)
    int k = idx & (CDIM - 1);   // output col  (0..1023)
    wqt[idx] = (_Float16)Wq[k * PROJC + c];
    wkt[idx] = (_Float16)Wk[k * PROJC + c];
}

// ---------------------------------------------------------------------------
// Projection GEMM: [4096,1024]f16 @ [1024,256] -> sq/sk [4096,256] f16
// One wave per 16x16 output tile; K-loop of v_wmma_f32_16x16x32_f16.
__global__ __launch_bounds__(32)
void lr_proj(const _Float16* __restrict__ xh,
             const _Float16* __restrict__ wqt, const _Float16* __restrict__ wkt,
             _Float16* __restrict__ sq, _Float16* __restrict__ sk) {
    const int rt = blockIdx.x;          // row tile  (0..255)
    const int ct = blockIdx.y;          // col tile  (0..15)
    const int z  = blockIdx.z;          // 0 = q, 1 = k
    const _Float16* wt = z ? wkt : wqt;
    _Float16* out      = z ? sk  : sq;

    const int lane = threadIdx.x;
    const int n = lane & 15;            // A: row-in-tile / B,D: col-in-tile
    const int g = lane >> 4;            // lane half

    // A lane m holds row m; K split: half0 -> {0..15}, half1 -> {16..31}
    const _Float16* arow = xh + (rt * 16 + n) * CDIM;
    // B lane n holds output column ct*16+n; K groups of 16 per lane half
    const _Float16* brow = wt + (ct * 16 + n) * CDIM + 16 * g;

    v8f acc = {};
    for (int kk = 0; kk < CDIM / 32; ++kk) {
        const int kbase = kk * 32;
        union { v16h v; v8h h[2]; } A;
        A.h[0] = *(const v8h*)(arow + kbase + 16 * g);
        A.h[1] = *(const v8h*)(arow + kbase + 16 * g + 8);
        v16h B = *(const v16h*)(brow + kbase);
        acc = __builtin_amdgcn_wmma_f32_16x16x32_f16(false, A.v, false, B,
                                                     (short)0, acc, false, false);
    }
    // D layout: VGPR v -> row (v + 8*g), lanes%16 -> col
    const int orow = rt * 16 + 8 * g;
    const int ocol = ct * 16 + n;
#pragma unroll
    for (int v = 0; v < 8; ++v)
        out[(orow + v) * PROJC + ocol] = (_Float16)acc[v];
}

// ---------------------------------------------------------------------------
// Fused: scores (WMMA) + causal/base mask + gumbel + row softmax + store.
// grid = (T/16, H, B); block = 512 threads = 16 waves, wave w owns cols [w*128, w*128+128)
__global__ __launch_bounds__(512)
void lr_scores_softmax(const _Float16* __restrict__ sq, const _Float16* __restrict__ sk,
                       const unsigned char* __restrict__ base_mask,
                       const float* __restrict__ gumbel, float* __restrict__ out) {
    const int it = blockIdx.x;          // i tile (0..127)
    const int h  = blockIdx.y;
    const int b  = blockIdx.z;
    const int lane = threadIdx.x & 31;
    const int w    = threadIdx.x >> 5;  // wave id (0..15)
    const int n = lane & 15;
    const int g = lane >> 4;
    const int i0 = it * 16;

    // A operand: sq rows i0..i0+15 for head h (loaded once)
    const _Float16* arow = sq + ((long)(b * TDIM + i0 + n)) * PROJC + h * RDIM;
    union { v16h v; v8h h2[2]; } A;
    A.h2[0] = *(const v8h*)(arow + 16 * g);
    A.h2[1] = *(const v8h*)(arow + 16 * g + 8);

    // 8 score tiles per wave: 16 rows x 128 cols
    v8f acc[8];
#pragma unroll
    for (int t = 0; t < 8; ++t) {
        const int j0 = w * 128 + t * 16;
        const _Float16* brow = sk + ((long)(b * TDIM + j0 + n)) * PROJC + h * RDIM + 16 * g;
        v16h B = *(const v16h*)brow;
        v8f c = {};
        acc[t] = __builtin_amdgcn_wmma_f32_16x16x32_f16(false, A.v, false, B,
                                                        (short)0, c, false, false);
    }

    const float scale = 0.17677669529663687f;  // 1/sqrt(RD)
    const float NEGINF = -__builtin_inff();
    const long  hb = ((long)(b * HNUM + h)) * TDIM;  // row base into [*,T] planes

    // logits = masked ? -inf : score*scale + gumbel ; track per-row partial max
    // gumbel is streamed exactly once (268 MB > L2) -> non-temporal loads.
    float pm[8];
#pragma unroll
    for (int v = 0; v < 8; ++v) {
        pm[v] = NEGINF;
        const int i = i0 + v + 8 * g;
        const float*         grow = gumbel    + (hb + i) * (long)TDIM;
        const unsigned char* mrow = base_mask + ((long)h * TDIM + i) * (long)TDIM;
#pragma unroll
        for (int t = 0; t < 8; ++t) {
            const int j = w * 128 + t * 16 + n;
            const float gv = __builtin_nontemporal_load(grow + j);
            const unsigned char mv = mrow[j];
            const bool masked = (j > i) || (mv != 0);
            const float val = masked ? NEGINF : fmaf(acc[t][v], scale, gv);
            acc[t][v] = val;
            pm[v] = fmaxf(pm[v], val);
        }
    }
    // reduce max across the 16 lanes of each half-wave (columns)
#pragma unroll
    for (int v = 0; v < 8; ++v)
        for (int m = 1; m < 16; m <<= 1)
            pm[v] = fmaxf(pm[v], __shfl_xor(pm[v], m, 32));

    __shared__ float redM[16 * 16];   // [row][wave]
    __shared__ float redS[16 * 16];
    if (n == 0) {
#pragma unroll
        for (int v = 0; v < 8; ++v) redM[(v + 8 * g) * 16 + w] = pm[v];
    }
    __syncthreads();

    float rmax[8];
#pragma unroll
    for (int v = 0; v < 8; ++v) {
        const int r = v + 8 * g;
        float m = redM[r * 16];
        for (int ww = 1; ww < 16; ++ww) m = fmaxf(m, redM[r * 16 + ww]);
        rmax[v] = m;
    }

    // exp + per-row partial sums
    float ps[8];
#pragma unroll
    for (int v = 0; v < 8; ++v) {
        float s = 0.0f;
#pragma unroll
        for (int t = 0; t < 8; ++t) {
            const float e = __expf(acc[t][v] - rmax[v]);
            acc[t][v] = e;
            s += e;
        }
        for (int m = 1; m < 16; m <<= 1) s += __shfl_xor(s, m, 32);
        ps[v] = s;
    }
    if (n == 0) {
#pragma unroll
        for (int v = 0; v < 8; ++v) redS[(v + 8 * g) * 16 + w] = ps[v];
    }
    __syncthreads();

    // normalize + store (268 MB streamed once -> non-temporal stores)
#pragma unroll
    for (int v = 0; v < 8; ++v) {
        const int r = v + 8 * g;
        float s = 0.0f;
        for (int ww = 0; ww < 16; ++ww) s += redS[r * 16 + ww];
        const float inv = 1.0f / s;
        const int i = i0 + r;
        float* orow = out + (hb + i) * (long)TDIM;
#pragma unroll
        for (int t = 0; t < 8; ++t) {
            const int j = w * 128 + t * 16 + n;
            __builtin_nontemporal_store(acc[t][v] * inv, orow + j);
        }
    }
}

// ---------------------------------------------------------------------------
extern "C" void kernel_launch(void* const* d_in, const int* in_sizes, int n_in,
                              void* d_out, int out_size, void* d_ws, size_t ws_size,
                              hipStream_t stream) {
    const float*         x         = (const float*)d_in[0];
    const unsigned char* base_mask = (const unsigned char*)d_in[1];
    const float*         gumbel    = (const float*)d_in[2];
    const float*         Wq        = (const float*)d_in[3];
    const float*         Wk        = (const float*)d_in[4];
    float* out = (float*)d_out;

    char* ws = (char*)d_ws;
    _Float16* xh  = (_Float16*)(ws);                                // 4096*1024*2 = 8 MB
    _Float16* wqt = (_Float16*)(ws + (size_t)8 * 1024 * 1024);      // 0.5 MB
    _Float16* wkt = (_Float16*)(ws + (size_t)8 * 1024 * 1024 + 524288);
    _Float16* sq  = (_Float16*)(ws + (size_t)9 * 1024 * 1024);      // 4096*256*2 = 2 MB
    _Float16* sk  = (_Float16*)(ws + (size_t)11 * 1024 * 1024);     // 2 MB

    // 1) cast x -> f16 (8 elems / thread)
    {
        const int n8 = (BROWS * CDIM) / 8;  // 512K
        lr_cast_x<<<(n8 + 255) / 256, 256, 0, stream>>>(x, xh, n8);
    }
    // 2) transpose+cast Wq, Wk
    {
        const int n = PROJC * CDIM;  // 256K
        lr_transpose_w<<<(n + 255) / 256, 256, 0, stream>>>(Wq, Wk, wqt, wkt);
    }
    // 3) projection GEMMs via WMMA
    {
        dim3 grid(BROWS / 16, PROJC / 16, 2);   // (256, 16, 2)
        lr_proj<<<grid, 32, 0, stream>>>(xh, wqt, wkt, sq, sk);
    }
    // 4) fused scores + mask + gumbel + softmax
    {
        dim3 grid(TDIM / 16, HNUM, 2);          // (128, 8, 2)
        lr_scores_softmax<<<grid, 512, 0, stream>>>(sq, sk, base_mask, gumbel, out);
    }
}
